// DeepSetJastrow_75067438399625
// MI455X (gfx1250) — matrix operand order repaired and checked by reading, not compiled, not run on hardware
//
#include <hip/hip_runtime.h>
#include <hip/hip_bf16.h>

typedef __attribute__((ext_vector_type(16))) _Float16 v16h;
typedef __attribute__((ext_vector_type(8)))  _Float16 v8h;
typedef __attribute__((ext_vector_type(8)))  float    v8f;

#define NPART 16
#define NPAIR 120

constexpr int   SH      = 136;                 // LDS row stride (halves); 272B = 16B-aligned, 4-bank skew
constexpr int   HHALVES = 128 * SH;
constexpr int   HBYTES  = HHALVES * 2;         // 34816 B
constexpr int   SLAB    = HHALVES;             // one weight slab in d_ws (halves), same layout as LDS

// LDS: H0 + H1 + two weight buffers + f32 scratch
constexpr size_t SMEM_BYTES = 4ull * HBYTES + (128 + 128 + 128 + 132 + 1024 + 1024 + 256 + 8) * 4;

struct Params {
    const float *x;
    const float *W1, *b1, *W2, *b2, *W3, *b3, *W4, *b4, *W5, *b5;
    const float *Wa0, *ba0, *Wa1, *ba1;
    const float *Ws1, *bs1, *Ws2, *bs2, *Ws3, *bs3;
    const float *Wr1, *br1, *Wr2, *br2, *Wr3, *br3, *Wr4, *br4;
};

// ---- branchless fast math ----
__device__ __forceinline__ float frcp(float x)  { return __builtin_amdgcn_rcpf(x); }
__device__ __forceinline__ float fexp(float x)  { return __expf(x); }
__device__ __forceinline__ float flog(float x)  { return __logf(x); }
__device__ __forceinline__ float gelu(float x) {
    float u = x * (1.595769122f + 0.071354816f * x * x);
    return x * frcp(1.0f + fexp(-u));
}
__device__ __forceinline__ float ftanh(float s) {
    return 1.0f - 2.0f * frcp(1.0f + fexp(2.0f * s));
}

// ---- CDNA5 async global->LDS staging (ASYNCcnt), 16B per lane per issue ----
__device__ __forceinline__ void stage_async(const _Float16* __restrict__ gsrc,
                                            _Float16* __restrict__ ldst,
                                            int bytes, int tid) {
    unsigned ldsBase = (unsigned)(uintptr_t)ldst;                 // low 32 bits of flat LDS addr = LDS offset
    unsigned long long gbase = (unsigned long long)(uintptr_t)gsrc;
    for (int c = tid * 16; c < bytes; c += 256 * 16) {
        asm volatile("global_load_async_to_lds_b128 %0, %1, %2"
                     :: "v"(ldsBase + (unsigned)c), "v"((unsigned)c), "s"(gbase)
                     : "memory");
    }
}
__device__ __forceinline__ void wait_async() {
    asm volatile("s_wait_asynccnt 0x0" ::: "memory");
}

// One 16x16 output tile: A (16 x 32*ksteps, rows of Hin) x B (cols of Wt), f32 accumulate.
__device__ __forceinline__ v8f wmma_tile(const _Float16* __restrict__ Hin,
                                         const _Float16* __restrict__ Wt,
                                         int mtile, int ntile, int ksteps,
                                         int lane, v8f acc) {
    const int mrow = mtile * 16 + (lane & 15);
    const int ncol = ntile * 16 + (lane & 15);
    const int half = lane >> 4;
    #pragma unroll
    for (int ks = 0; ks < 4; ++ks) {
        if (ks >= ksteps) break;
        const int kb = ks * 32;
        v8h alo = *(const v8h*)(Hin + mrow * SH + kb + half * 8);
        v8h ahi = *(const v8h*)(Hin + mrow * SH + kb + 16 + half * 8);
        v8h blo = *(const v8h*)(Wt + ncol * SH + kb + half * 8);
        v8h bhi = *(const v8h*)(Wt + ncol * SH + kb + 16 + half * 8);
        v16h a, b;
        #pragma unroll
        for (int i = 0; i < 8; ++i) {
            a[i] = alo[i]; a[8 + i] = ahi[i];
            b[i] = blo[i]; b[8 + i] = bhi[i];
        }
        acc = __builtin_amdgcn_wmma_f32_16x16x32_f16(false, a, false, b,
                                                     (short)0, acc, false, false);
    }
    return acc;
}

// ---- prep: transpose+convert pair-MLP weights into d_ws slabs, f16, [n][SH] layout ----
// slab s (s=0..4) holds weights of pair-MLP layer s+1; layer1 K-padded to 32 rows (13 valid).
__global__ __launch_bounds__(256) void prep_weights(Params p, _Float16* __restrict__ wt) {
    int idx = blockIdx.x * 256 + threadIdx.x;
    if (idx >= 5 * 128 * 128) return;
    int l = idx / (128 * 128);
    int r = idx - l * (128 * 128);
    int n = r >> 7;
    int k = r & 127;
    float v = 0.0f;
    if (l == 0)      { if (k < 13) v = p.W1[k * 128 + n]; }
    else if (l == 1) { v = p.W2[k * 128 + n]; }
    else if (l == 2) { v = p.W3[k * 128 + n]; }
    else if (l == 3) { v = p.W4[k * 128 + n]; }
    else             { if (n < 32) v = p.W5[k * 32 + n]; else v = 0.0f; }
    wt[(size_t)l * SLAB + n * SH + k] = (_Float16)v;
}

__global__ __launch_bounds__(256) void jastrow_kernel(Params p,
                                                      const _Float16* __restrict__ wt,
                                                      float* __restrict__ out) {
    extern __shared__ char smem[];
    _Float16* H0 = (_Float16*)smem;
    _Float16* H1 = H0 + HHALVES;
    _Float16* Wb0 = H1 + HHALVES;
    _Float16* Wb1 = Wb0 + HHALVES;
    float* biasL = (float*)(Wb1 + HHALVES);
    float* gateL = biasL + 128;
    float* logit = gateL + 128;
    float* rho   = logit + 128;
    float* sp1   = rho + 132;
    float* sp2   = sp1 + 1024;
    float* spe   = sp2 + 1024;
    float* scal  = spe + 256;                // [0]=s1sum [1]=cusp [2]=r2sum [3]=expsum [4]=maxlogit
    float* embF  = (float*)H1;               // layer-5 output (120x32 f32), reuses pong buffer
    _Float16* Wbuf[2] = { Wb0, Wb1 };

    const int tid  = threadIdx.x;
    const int lane = tid & 31;
    const int wave = tid >> 5;
    const int bidx = blockIdx.x;
    const float* xb = p.x + bidx * (NPART * 3);

    constexpr float F32_EPS = 1.1920929e-07f;
    constexpr float EPS2    = 0.04f;
    constexpr float GATE2   = 0.09f;

    // ---- kick off layer-1 weight DMA immediately; it overlaps feature construction ----
    stage_async(wt, Wb0, HBYTES, tid);

    // ---- init ----
    if (tid < 8) scal[tid] = 0.0f;
    for (int i = tid; i < 128; i += 256) gateL[i] = 0.0f;
    for (int i = tid; i < 128 * 32; i += 256) {
        int r = i >> 5, c = i & 31;
        H0[r * SH + c] = (_Float16)0.0f;
    }
    __syncthreads();

    // ---- pair features, cusp, s1 sum (threads 0..119) ----
    if (tid < NPAIR) {
        int pi = 0, rem = tid;
        while (rem >= (NPART - 1 - pi)) { rem -= (NPART - 1 - pi); ++pi; }
        int pj = pi + 1 + rem;
        float xi0 = xb[pi*3+0], xi1 = xb[pi*3+1], xi2 = xb[pi*3+2];
        float xj0 = xb[pj*3+0], xj1 = xb[pj*3+1], xj2 = xb[pj*3+2];
        float d0 = xi0 - xj0, d1 = xi1 - xj1, d2 = xi2 - xj2;
        float c0 = 0.5f*(xi0+xj0), c1 = 0.5f*(xi1+xj1), c2 = 0.5f*(xi2+xj2);
        float r2  = d0*d0 + d1*d1 + d2*d2;
        float r   = __builtin_amdgcn_sqrtf(r2 + F32_EPS);
        float rr2 = r * r;
        float rt2 = (rr2 + EPS2) * (1.0f / EPS2);
        float s1  = flog(1.0f + rt2);
        float s2  = rr2 * frcp(rr2 + EPS2);
        float s3  = rt2 * fexp(-rt2);
        float rb0 = fexp(-0.25f * s1), rb1 = fexp(-s1), rb2 = fexp(-4.0f * s1);
        float sm  = ((pi < 8) == (pj < 8)) ? 1.0f : 0.0f;
        _Float16* row = H0 + tid * SH;
        row[0]=(_Float16)s1; row[1]=(_Float16)s2; row[2]=(_Float16)s3;
        row[3]=(_Float16)rb0; row[4]=(_Float16)rb1; row[5]=(_Float16)rb2;
        row[6]=(_Float16)d0; row[7]=(_Float16)d1; row[8]=(_Float16)d2;
        row[9]=(_Float16)c0; row[10]=(_Float16)c1; row[11]=(_Float16)c2;
        row[12]=(_Float16)sm;
        gateL[tid] = rr2 * frcp(rr2 + GATE2);
        atomicAdd(&scal[0], s1);
        float rc  = __builtin_amdgcn_sqrtf(r2 + 1e-30f);
        float gam = (sm > 0.5f) ? 0.25f : 0.5f;
        atomicAdd(&scal[1], gam * rc * fexp(-rc));
    }
    if (tid < 48) { float xv = xb[tid]; atomicAdd(&scal[2], xv * xv); }
    wait_async();                      // layer-1 weights landed (per-wave), barrier makes global
    __syncthreads();

    const float* biases[5] = { p.b1, p.b2, p.b3, p.b4, p.b5 };
    const int half = lane >> 4, nc = lane & 15;

    // ---- pair MLP layer 1: K=32 (13 valid), N=128, H0 -> H1, weights in Wb0 ----
    for (int i = tid; i < 128; i += 256) biasL[i] = biases[0][i];
    stage_async(wt + 1 * SLAB, Wb1, HBYTES, tid);     // layer-2 weights, overlapped with L1 compute
    __syncthreads();
    for (int nt = 0; nt < 8; ++nt) {
        float bv = biasL[nt * 16 + nc];
        v8f acc;
        #pragma unroll
        for (int r = 0; r < 8; ++r) acc[r] = bv;
        acc = wmma_tile(H0, Wb0, wave, nt, 1, lane, acc);
        #pragma unroll
        for (int r = 0; r < 8; ++r) {
            int row = wave * 16 + half * 8 + r;
            H1[row * SH + nt * 16 + nc] = (_Float16)gelu(acc[r]);
        }
    }
    __syncthreads();
    wait_async();
    __syncthreads();

    // ---- pair MLP layers 2..4: 128x128, ping-pong activations, double-buffered weights ----
    for (int j = 0; j < 3; ++j) {
        const _Float16* Hin  = (j & 1) ? H0 : H1;
        _Float16*       Hout = (j & 1) ? H1 : H0;
        const int cur = (j + 1) & 1;                   // L2:Wb1, L3:Wb0, L4:Wb1
        for (int i = tid; i < 128; i += 256) biasL[i] = biases[j + 1][i];
        stage_async(wt + (size_t)(j + 2) * SLAB, Wbuf[cur ^ 1],
                    (j + 2 == 4) ? (32 * SH * 2) : HBYTES, tid);   // next slab (L5 is 32 rows)
        __syncthreads();
        for (int nt = 0; nt < 8; ++nt) {
            float bv = biasL[nt * 16 + nc];
            v8f acc;
            #pragma unroll
            for (int r = 0; r < 8; ++r) acc[r] = bv;
            acc = wmma_tile(Hin, Wbuf[cur], wave, nt, 4, lane, acc);
            #pragma unroll
            for (int r = 0; r < 8; ++r) {
                int row = wave * 16 + half * 8 + r;
                Hout[row * SH + nt * 16 + nc] = (_Float16)gelu(acc[r]);
            }
        }
        __syncthreads();
        wait_async();
        __syncthreads();
    }

    // ---- pair MLP layer 5: K=128, N=32, linear, *gate; H0 -> embF; weights in Wb0 ----
    {
        for (int i = tid; i < 32; i += 256) biasL[i] = biases[4][i];
        __syncthreads();
        float myGate[8];
        #pragma unroll
        for (int r = 0; r < 8; ++r) {
            int row = wave * 16 + half * 8 + r;
            myGate[r] = gateL[row];
        }
        v8f accT[2];
        for (int nt = 0; nt < 2; ++nt) {
            float bv = biasL[nt * 16 + nc];
            v8f acc;
            #pragma unroll
            for (int r = 0; r < 8; ++r) acc[r] = bv;
            accT[nt] = wmma_tile(H0, Wb0, wave, nt, 4, lane, acc);
        }
        __syncthreads();
        for (int nt = 0; nt < 2; ++nt) {
            #pragma unroll
            for (int r = 0; r < 8; ++r) {
                int row = wave * 16 + half * 8 + r;
                embF[row * 32 + nt * 16 + nc] = accT[nt][r] * myGate[r];
            }
        }
    }
    __syncthreads();

    // ---- attention logits (threads 0..119) ----
    if (tid < NPAIR) {
        const float* e = embF + tid * 32;
        float lg = p.ba1[0];
        for (int u = 0; u < 16; ++u) {
            float s = p.ba0[u];
            for (int c = 0; c < 32; ++c) s += e[c] * p.Wa0[c * 16 + u];
            lg += ftanh(s) * p.Wa1[u];
        }
        logit[tid] = lg;
    }
    __syncthreads();
    if (tid == 0) {
        float m = -1e30f;
        for (int i = 0; i < NPAIR; ++i) m = fmaxf(m, logit[i]);
        scal[4] = m;
    }
    __syncthreads();
    if (tid < NPAIR) {
        float w = fexp(logit[tid] - scal[4]);
        logit[tid] = w;
        atomicAdd(&scal[3], w);
    }
    __syncthreads();

    // ---- pair reductions -> rho[0..95] ----
    if (tid < 32) {
        float s = 0.0f, aw = 0.0f;
        for (int pp = 0; pp < NPAIR; ++pp) {
            float v = embF[pp * 32 + tid];
            s += v; aw += v * logit[pp];
        }
        rho[tid]      = s;
        rho[32 + tid] = s * (1.0f / NPAIR);
        rho[64 + tid] = aw * frcp(scal[3]);
    }

    // ---- single-particle MLP: 16 x (4 -> 64 -> 64 -> 16) ----
    for (int idx = tid; idx < 16 * 64; idx += 256) {
        int t = idx >> 6, u = idx & 63;
        const float* xt = xb + t * 3;
        float sp = (t < 8) ? 0.0f : 1.0f;
        float s = p.bs1[u] + xt[0]*p.Ws1[u] + xt[1]*p.Ws1[64+u] + xt[2]*p.Ws1[128+u] + sp*p.Ws1[192+u];
        sp1[idx] = gelu(s);
    }
    __syncthreads();
    for (int idx = tid; idx < 16 * 64; idx += 256) {
        int t = idx >> 6, u = idx & 63;
        float s = p.bs2[u];
        for (int k = 0; k < 64; ++k) s += sp1[t * 64 + k] * p.Ws2[k * 64 + u];
        sp2[idx] = gelu(s);
    }
    __syncthreads();
    for (int idx = tid; idx < 16 * 16; idx += 256) {
        int t = idx >> 4, u = idx & 15;
        float s = p.bs3[u];
        for (int k = 0; k < 64; ++k) s += sp2[t * 64 + k] * p.Ws3[k * 16 + u];
        spe[idx] = s;
    }
    __syncthreads();
    if (tid < 16) {
        float s = 0.0f;
        for (int t = 0; t < 16; ++t) s += spe[t * 16 + tid];
        rho[96 + tid]  = s;
        rho[112 + tid] = s * (1.0f / 16.0f);
    }
    if (tid == 0) {
        rho[128] = scal[2] * (1.0f / 48.0f);   // r2_mean
        rho[129] = scal[0] * (1.0f / NPAIR);   // s1_mean
    }
    __syncthreads();

    // ---- readout MLP: 130 -> 128 -> 128 -> 128 -> 1 (GEMV, 128 threads) ----
    float* hA = sp1; float* hB = sp2;
    if (tid < 128) {
        float s = p.br1[tid];
        for (int k = 0; k < 130; ++k) s += rho[k] * p.Wr1[k * 128 + tid];
        hA[tid] = gelu(s);
    }
    __syncthreads();
    if (tid < 128) {
        float s = p.br2[tid];
        for (int k = 0; k < 128; ++k) s += hA[k] * p.Wr2[k * 128 + tid];
        hB[tid] = gelu(s);
    }
    __syncthreads();
    if (tid < 128) {
        float s = p.br3[tid];
        for (int k = 0; k < 128; ++k) s += hB[k] * p.Wr3[k * 128 + tid];
        hA[tid] = gelu(s);
    }
    __syncthreads();
    if (tid == 0) {
        float s = p.br4[0];
        for (int k = 0; k < 128; ++k) s += hA[k] * p.Wr4[k];
        out[bidx] = s + scal[1];               // f_nn + cusp
    }
}

extern "C" void kernel_launch(void* const* d_in, const int* in_sizes, int n_in,
                              void* d_out, int out_size, void* d_ws, size_t ws_size,
                              hipStream_t stream) {
    Params p;
    p.x   = (const float*)d_in[0];
    p.W1  = (const float*)d_in[1];  p.b1  = (const float*)d_in[2];
    p.W2  = (const float*)d_in[3];  p.b2  = (const float*)d_in[4];
    p.W3  = (const float*)d_in[5];  p.b3  = (const float*)d_in[6];
    p.W4  = (const float*)d_in[7];  p.b4  = (const float*)d_in[8];
    p.W5  = (const float*)d_in[9];  p.b5  = (const float*)d_in[10];
    p.Wa0 = (const float*)d_in[11]; p.ba0 = (const float*)d_in[12];
    p.Wa1 = (const float*)d_in[13]; p.ba1 = (const float*)d_in[14];
    p.Ws1 = (const float*)d_in[15]; p.bs1 = (const float*)d_in[16];
    p.Ws2 = (const float*)d_in[17]; p.bs2 = (const float*)d_in[18];
    p.Ws3 = (const float*)d_in[19]; p.bs3 = (const float*)d_in[20];
    p.Wr1 = (const float*)d_in[21]; p.br1 = (const float*)d_in[22];
    p.Wr2 = (const float*)d_in[23]; p.br2 = (const float*)d_in[24];
    p.Wr3 = (const float*)d_in[25]; p.br3 = (const float*)d_in[26];
    p.Wr4 = (const float*)d_in[27]; p.br4 = (const float*)d_in[28];

    _Float16* wt = (_Float16*)d_ws;                    // 5 slabs * 34816 B = 174080 B
    const int B = in_sizes[0] / (NPART * 3);

    prep_weights<<<dim3((5 * 128 * 128 + 255) / 256), dim3(256), 0, stream>>>(p, wt);
    jastrow_kernel<<<dim3(B), dim3(256), SMEM_BYTES, stream>>>(p, wt, (float*)d_out);
}
